// ClauseToVarLayer_13597866459550
// MI455X (gfx1250) — matrix operand order repaired and checked by reading, not compiled, not run on hardware
//
#include <hip/hip_runtime.h>
#include <hip/hip_bf16.h>
#include <stdint.h>

// CDNA5 / gfx1250, wave32. D = 128 fixed by the model.
#define DD    128
#define KTOT  256          // concatenated K: msg(128) + h(128)
#define NG    512          // 4*D gate columns
#define LDSW  520          // padded LDS row stride (floats) for gate staging

typedef __bf16 v16bf __attribute__((ext_vector_type(16)));
typedef float  v8f   __attribute__((ext_vector_type(8)));
typedef unsigned int u32x4 __attribute__((ext_vector_type(4)));

union BF16Frag {
    v16bf v;
    u32x4 q[2];
};

// ---------------------------------------------------------------- zero msg
__global__ void zero_f4_kernel(float4* __restrict__ p, int n4) {
    int i = blockIdx.x * blockDim.x + threadIdx.x;
    if (i < n4) p[i] = make_float4(0.f, 0.f, 0.f, 0.f);
}

// ---------------------------------------------------------------- scatter-add
// One wave per edge; lane l handles d = l*4 .. l*4+3 (float4 gather, 4 f32 atomics).
__global__ void scatter_kernel(const int* __restrict__ ec, const int* __restrict__ ev,
                               const float* __restrict__ xc, float* __restrict__ msg,
                               int n_edges) {
    int gid  = blockIdx.x * blockDim.x + threadIdx.x;
    int edge = gid >> 5;
    int lane = gid & 31;
    if (edge >= n_edges) return;
    int cl = ec[edge];
    int vr = ev[edge];
    float4 x = reinterpret_cast<const float4*>(xc + (size_t)cl * DD)[lane];
    float* dst = msg + (size_t)vr * DD + lane * 4;
    __hip_atomic_fetch_add(dst + 0, x.x, __ATOMIC_RELAXED, __HIP_MEMORY_SCOPE_AGENT);
    __hip_atomic_fetch_add(dst + 1, x.y, __ATOMIC_RELAXED, __HIP_MEMORY_SCOPE_AGENT);
    __hip_atomic_fetch_add(dst + 2, x.z, __ATOMIC_RELAXED, __HIP_MEMORY_SCOPE_AGENT);
    __hip_atomic_fetch_add(dst + 3, x.w, __ATOMIC_RELAXED, __HIP_MEMORY_SCOPE_AGENT);
}

// ---------------------------------------------------------------- pack A = [msg | h] -> bf16
__global__ void pack_msgh_kernel(const float* __restrict__ msg, const float* __restrict__ h,
                                 __bf16* __restrict__ msgh, int total) {
    int i = blockIdx.x * blockDim.x + threadIdx.x;
    if (i >= total) return;
    int col = i & (KTOT - 1);
    int row = i >> 8;                  // KTOT == 256
    float v = (col < DD) ? msg[(size_t)row * DD + col]
                         : h[(size_t)row * DD + (col - DD)];
    msgh[i] = (__bf16)v;
}

// ---------------------------------------------------------------- pack W = [W_ih | W_hh] -> bf16
__global__ void pack_w_kernel(const float* __restrict__ Wih, const float* __restrict__ Whh,
                              __bf16* __restrict__ wcat, int total) {
    int i = blockIdx.x * blockDim.x + threadIdx.x;
    if (i >= total) return;
    int col = i & (KTOT - 1);
    int row = i >> 8;
    float v = (col < DD) ? Wih[(size_t)row * DD + col]
                         : Whh[(size_t)row * DD + (col - DD)];
    wcat[i] = (__bf16)v;
}

// ---------------------------------------------------------------- fused WMMA GEMM + LSTM cell
// Block = 256 threads = 8 waves. Block computes rows [mbase, mbase+16) x all 512 gates.
// Wave w owns gate columns [w*64, w*64+64) as 4 accumulator tiles of 16x16 f32.
// Per K-step: all A/B fragments are issued as a batch of distinct live values so the
// compiler keeps 8+ loads in flight and staggers s_wait_loadcnt across the 4 WMMAs.
__global__ __launch_bounds__(256) void lstm_gemm_kernel(
    const __bf16* __restrict__ msgh,   // [n_vars, 256] row-major
    const __bf16* __restrict__ wcat,   // [512, 256]    row-major (along K)
    const float*  __restrict__ b_ih,   // [512]
    const float*  __restrict__ b_hh,   // [512]
    const float*  __restrict__ c_old,  // [n_vars, 128]
    float* __restrict__ out_h,         // [n_vars, 128]
    float* __restrict__ out_c,         // [n_vars, 128]
    int n_vars) {
    __shared__ float gat[16 * LDSW];

    const int tid   = threadIdx.x;
    const int lane  = tid & 31;
    const int wave  = tid >> 5;            // 0..7
    const int mbase = blockIdx.x * 16;
    const int nbase = wave * 64;
    const int l16   = lane & 15;
    const int lhi   = lane >> 4;           // lane half: 0 or 1

    v8f acc[4] = {};

    // A fragment base for this lane (ISA 16-bit A 16x32 layout: lane = M).
    // lanes 0-15: K = k0+0..7 then k0+16..23 ; lanes 16-31: K = k0+8..15 then k0+24..31
    const __bf16* arow = msgh + (size_t)(mbase + l16) * KTOT + lhi * 8;
    // B fragment base (B 32x16 bf16 layout: lane = N mod 16, same K order as A).
    const __bf16* brow = wcat + (size_t)(nbase + l16) * KTOT + lhi * 8;

#pragma unroll
    for (int k0 = 0; k0 < KTOT; k0 += 32) {
        BF16Frag a;
        a.q[0] = *reinterpret_cast<const u32x4*>(arow + k0);
        a.q[1] = *reinterpret_cast<const u32x4*>(arow + k0 + 16);

        BF16Frag b[4];
#pragma unroll
        for (int t = 0; t < 4; ++t) {
            const __bf16* bp = brow + (size_t)t * 16 * KTOT + k0;
            b[t].q[0] = *reinterpret_cast<const u32x4*>(bp);
            b[t].q[1] = *reinterpret_cast<const u32x4*>(bp + 16);
        }
#pragma unroll
        for (int t = 0; t < 4; ++t) {
            acc[t] = __builtin_amdgcn_wmma_f32_16x16x32_bf16(
                false, a.v, false, b[t].v, (short)0, acc[t], false, false);
        }
    }

    // Stage gates in LDS (C/D layout: VGPR r -> M = r + 8*lane_half, N = lane%16).
#pragma unroll
    for (int t = 0; t < 4; ++t) {
        const int n = nbase + t * 16 + l16;
#pragma unroll
        for (int r = 0; r < 8; ++r) {
            const int m = r + lhi * 8;
            gat[m * LDSW + n] = acc[t][r];
        }
    }
    __syncthreads();

    // Elementwise LSTM: thread t -> row m = t/16, d-slice of 8.
    const int m     = tid >> 4;
    const int dbase = (tid & 15) * 8;
    const int var   = mbase + m;
    if (var < n_vars) {
#pragma unroll
        for (int j = 0; j < 8; ++j) {
            const int d = dbase + j;
            float gi = gat[m * LDSW + d]           + b_ih[d]           + b_hh[d];
            float gf = gat[m * LDSW + DD + d]      + b_ih[DD + d]      + b_hh[DD + d];
            float gg = gat[m * LDSW + 2 * DD + d]  + b_ih[2 * DD + d]  + b_hh[2 * DD + d];
            float go = gat[m * LDSW + 3 * DD + d]  + b_ih[3 * DD + d]  + b_hh[3 * DD + d];
            float ig = 1.f / (1.f + __expf(-gi));
            float fg = 1.f / (1.f + __expf(-gf));
            float gv = tanhf(gg);
            float og = 1.f / (1.f + __expf(-go));
            float cn = fg * c_old[(size_t)var * DD + d] + ig * gv;
            out_h[(size_t)var * DD + d] = og * tanhf(cn);
            out_c[(size_t)var * DD + d] = cn;
        }
    }
}

// ---------------------------------------------------------------- launcher
extern "C" void kernel_launch(void* const* d_in, const int* in_sizes, int n_in,
                              void* d_out, int out_size, void* d_ws, size_t ws_size,
                              hipStream_t stream) {
    const int*   ec  = (const int*)d_in[0];
    const int*   ev  = (const int*)d_in[1];
    const float* xc  = (const float*)d_in[2];
    const float* h   = (const float*)d_in[3];
    const float* c   = (const float*)d_in[4];
    const float* Wih = (const float*)d_in[5];
    const float* Whh = (const float*)d_in[6];
    const float* bih = (const float*)d_in[7];
    const float* bhh = (const float*)d_in[8];

    const int n_edges = in_sizes[0];
    const int n_vars  = in_sizes[3] / DD;

    float* out_h = (float*)d_out;
    float* out_c = out_h + (size_t)n_vars * DD;

    // Workspace carve-up (256B aligned): msg f32, A bf16, W bf16.
    char*   ws   = (char*)d_ws;
    float*  msg  = (float*)ws;
    size_t  off  = (((size_t)n_vars * DD * sizeof(float)) + 255) & ~(size_t)255;
    __bf16* msgh = (__bf16*)(ws + off);
    off += (((size_t)n_vars * KTOT * sizeof(__bf16)) + 255) & ~(size_t)255;
    __bf16* wcat = (__bf16*)(ws + off);

    // 1) zero msg
    {
        int n4 = n_vars * DD / 4;
        zero_f4_kernel<<<(n4 + 255) / 256, 256, 0, stream>>>((float4*)msg, n4);
    }
    // 2) scatter-add x_c onto msg (wave per edge)
    {
        long long thr = (long long)n_edges * 32;
        int blocks = (int)((thr + 255) / 256);
        scatter_kernel<<<blocks, 256, 0, stream>>>(ec, ev, xc, msg, n_edges);
    }
    // 3) pack A and W to bf16
    {
        int totalA = n_vars * KTOT;
        pack_msgh_kernel<<<(totalA + 255) / 256, 256, 0, stream>>>(msg, h, msgh, totalA);
        int totalW = NG * KTOT;
        pack_w_kernel<<<(totalW + 255) / 256, 256, 0, stream>>>(Wih, Whh, wcat, totalW);
    }
    // 4) fused WMMA GEMM + LSTM cell
    {
        int blocks = (n_vars + 15) / 16;
        lstm_gemm_kernel<<<blocks, 256, 0, stream>>>(msgh, wcat, bih, bhh, c,
                                                     out_h, out_c, n_vars);
    }
}
